// GATLayer_88089779241309
// MI455X (gfx1250) — matrix operand order repaired and checked by reading, not compiled, not run on hardware
//
#include <hip/hip_runtime.h>

typedef __attribute__((ext_vector_type(16))) __bf16 v16bf;
typedef __attribute__((ext_vector_type(8)))  __bf16 v8bf;
typedef __attribute__((ext_vector_type(8)))  float  v8f;
typedef __attribute__((ext_vector_type(4)))  int    v4i;

#define WMMA_BF16(A, B, Cacc) \
  __builtin_amdgcn_wmma_f32_16x16x32_bf16(false, (A), false, (B), (short)0, (Cacc), false, false)

static __device__ __forceinline__ v16bf cat16(v8bf lo, v8bf hi) {
  return __builtin_shufflevector(lo, hi, 0,1,2,3,4,5,6,7,8,9,10,11,12,13,14,15);
}
static __device__ __forceinline__ float lrelu(float x) { return x > 0.f ? x : 0.3f * x; }

// Async 16-byte global->LDS copy (gfx1250 ASYNC path, tracked by ASYNCcnt).
static __device__ __forceinline__ void async_cp16(const __bf16* g, __bf16* l) {
  typedef v4i __attribute__((address_space(1)))* gas_t;
  typedef v4i __attribute__((address_space(3)))* las_t;
  gas_t gp = (gas_t)(uintptr_t)g;                       // global: value-preserving
  las_t lp = (las_t)(uint32_t)(uintptr_t)l;             // LDS offset = low 32 bits
#if __has_builtin(__builtin_amdgcn_global_load_async_to_lds_b128)
  __builtin_amdgcn_global_load_async_to_lds_b128(gp, lp, 0, 0);
#else
  const uint32_t lo32 = (uint32_t)(uintptr_t)l;
  asm volatile("global_load_async_to_lds_b128 %0, %1, off"
               :: "v"(lo32), "v"(g) : "memory");
#endif
}

// ---------------------------------------------------------------------------
// Kernel 1: per 16-row tile: row norms, nf (bf16), h = x*W^T + b via WMMA,
//           s_l/s_r projections, h^T (bf16) for the PV GEMM.
// ---------------------------------------------------------------------------
__global__ __launch_bounds__(32) void gat_prep(
    const float* __restrict__ x, const float* __restrict__ W,
    const float* __restrict__ bias, const float* __restrict__ a,
    __bf16* __restrict__ nf, __bf16* __restrict__ hT,
    float* __restrict__ sl, float* __restrict__ sr)
{
  constexpr int N = 2048, C = 128;
  const int lane = threadIdx.x;
  const int row  = lane & 15;
  const int half = lane >> 4;
  const int b  = blockIdx.x / (N / 16);
  const int n0 = (blockIdx.x % (N / 16)) * 16;

  const float* xrow = x + ((size_t)(b * N + n0 + row)) * C;

  float fx[4][16];
  float ss = 0.f;
#pragma unroll
  for (int kc = 0; kc < 4; ++kc) {
    const int base = kc * 32 + half * 8;
    const float4 f0 = *(const float4*)(xrow + base);
    const float4 f1 = *(const float4*)(xrow + base + 4);
    const float4 f2 = *(const float4*)(xrow + base + 16);
    const float4 f3 = *(const float4*)(xrow + base + 20);
    fx[kc][0]=f0.x; fx[kc][1]=f0.y; fx[kc][2]=f0.z;  fx[kc][3]=f0.w;
    fx[kc][4]=f1.x; fx[kc][5]=f1.y; fx[kc][6]=f1.z;  fx[kc][7]=f1.w;
    fx[kc][8]=f2.x; fx[kc][9]=f2.y; fx[kc][10]=f2.z; fx[kc][11]=f2.w;
    fx[kc][12]=f3.x; fx[kc][13]=f3.y; fx[kc][14]=f3.z; fx[kc][15]=f3.w;
#pragma unroll
    for (int e = 0; e < 16; ++e) ss += fx[kc][e] * fx[kc][e];
  }
  ss += __shfl_xor(ss, 16, 32);                    // lanes l / l+16 cover one row
  const float inv = 1.0f / fmaxf(sqrtf(ss), 1e-12f);

  __bf16* nfrow = nf + ((size_t)(b * N + n0 + row)) * C;
  v16bf af[4];
#pragma unroll
  for (int kc = 0; kc < 4; ++kc) {
    const int base = kc * 32 + half * 8;
    v8bf lo, hi; v16bf t;
#pragma unroll
    for (int e = 0; e < 8; ++e) {
      lo[e]    = (__bf16)(fx[kc][e] * inv);
      hi[e]    = (__bf16)(fx[kc][e + 8] * inv);
      t[e]     = (__bf16)fx[kc][e];
      t[e + 8] = (__bf16)fx[kc][e + 8];
    }
    *(v8bf*)(nfrow + base)      = lo;
    *(v8bf*)(nfrow + base + 16) = hi;
    af[kc] = t;
  }

  // h tile (16 x 128) = A(16x128) x W^T, 8 col-tiles x 4 K-chunks of WMMA.
  v8f acc[8];
#pragma unroll
  for (int ct = 0; ct < 8; ++ct) {
    v8f cacc = {};
#pragma unroll
    for (int kc = 0; kc < 4; ++kc) {
      const float* wp = W + (size_t)(ct * 16 + row) * C + kc * 32 + half * 8;
      v16bf bfrag;
#pragma unroll
      for (int e = 0; e < 16; ++e) bfrag[e] = (__bf16)wp[e + (e & 8)];
      cacc = WMMA_BF16(af[kc], bfrag, cacc);
    }
    const float bv = bias[ct * 16 + row];
#pragma unroll
    for (int r = 0; r < 8; ++r) cacc[r] += bv;
    acc[ct] = cacc;
  }

  // s_l / s_r projections, reduced across the 16-lane group.
  float pl[4][8], pr[4][8];
#pragma unroll
  for (int hh = 0; hh < 4; ++hh)
#pragma unroll
    for (int r = 0; r < 8; ++r) { pl[hh][r] = 0.f; pr[hh][r] = 0.f; }
#pragma unroll
  for (int ct = 0; ct < 8; ++ct) {
    const int hh  = ct >> 1;
    const int c32 = (ct & 1) * 16 + row;
    const float wl = a[hh * 64 + c32];
    const float wr = a[hh * 64 + 32 + c32];
#pragma unroll
    for (int r = 0; r < 8; ++r) {
      pl[hh][r] += acc[ct][r] * wl;
      pr[hh][r] += acc[ct][r] * wr;
    }
  }
#pragma unroll
  for (int m = 1; m <= 8; m <<= 1)
#pragma unroll
    for (int hh = 0; hh < 4; ++hh)
#pragma unroll
      for (int r = 0; r < 8; ++r) {
        pl[hh][r] += __shfl_xor(pl[hh][r], m, 32);
        pr[hh][r] += __shfl_xor(pr[hh][r], m, 32);
      }
  if (row == 0) {
#pragma unroll
    for (int r = 0; r < 8; ++r) {
      const size_t n = (size_t)(b * N + n0 + half * 8 + r);
#pragma unroll
      for (int hh = 0; hh < 4; ++hh) {
        sl[n * 4 + hh] = pl[hh][r];
        sr[n * 4 + hh] = pr[hh][r];
      }
    }
  }

  // h^T (bf16): row = channel, col = node.
#pragma unroll
  for (int ct = 0; ct < 8; ++ct)
#pragma unroll
    for (int r = 0; r < 8; ++r) {
      const int c = ct * 16 + row;
      const int i = n0 + half * 8 + r;
      hT[((size_t)b * C + c) * N + i] = (__bf16)acc[ct][r];
    }
}

// ---------------------------------------------------------------------------
// Kernel 2: fused masked-softmax attention. 4 waves/block share one batch's
// j-panels, staged into LDS with double-buffered ASYNC global->LDS copies.
// Static softmax shift m = LR(s_l + max_j s_r) -> no accumulator rescaling.
// ---------------------------------------------------------------------------
__global__ __launch_bounds__(128) void gat_attn(
    const __bf16* __restrict__ nf, const __bf16* __restrict__ hT,
    const float* __restrict__ sl, const float* __restrict__ sr,
    float* __restrict__ out)
{
  constexpr int N = 2048, C = 128;
  const int tid  = threadIdx.x;
  const int wv   = tid >> 5;        // wave in block: 4 query tiles
  const int lane = tid & 31;
  const int row  = lane & 15;
  const int half = (lane >> 4) & 1;
  const int b  = blockIdx.x >> 5;                       // 32 blocks per batch
  const int i0 = (((blockIdx.x & 31) << 2) + wv) * 16;

  __shared__ __attribute__((aligned(16))) __bf16 nfbuf[2][32 * 128];  // j-panel of nf
  __shared__ __attribute__((aligned(16))) __bf16 htbuf[2][128 * 32];  // j-panel of h^T
  __shared__ __attribute__((aligned(16))) __bf16 stage[4][4][16 * 32];

  const __bf16* nfb = nf + (size_t)b * N * C;
  const __bf16* hTb = hT + (size_t)b * C * N;
  const float*  srb = sr + (size_t)b * N * 4;

  // Issue one 32-j panel (8 KB nf + 8 KB h^T) as 8 async b128 copies / wave.
  auto issue = [&](int j0n, int buf) {
#pragma unroll
    for (int it = 0; it < 4; ++it) {
      const int idx = it * 128 + tid;                   // 512 x 16B chunks
      async_cp16(nfb + (size_t)(j0n + (idx >> 4)) * C + (idx & 15) * 8,
                 &nfbuf[buf][(idx >> 4) * 128 + (idx & 15) * 8]);
    }
#pragma unroll
    for (int it = 0; it < 4; ++it) {
      const int idx = it * 128 + tid;
      async_cp16(hTb + (size_t)(idx >> 2) * N + j0n + (idx & 3) * 8,
                 &htbuf[buf][(idx >> 2) * 32 + (idx & 3) * 8]);
    }
  };

  // Per-head global max of s_r (upper-bound softmax shift).
  float M4[4] = {-3.0e38f, -3.0e38f, -3.0e38f, -3.0e38f};
  for (int j = lane; j < N; j += 32) {
    const float4 v = *(const float4*)(srb + (size_t)j * 4);
    M4[0] = fmaxf(M4[0], v.x); M4[1] = fmaxf(M4[1], v.y);
    M4[2] = fmaxf(M4[2], v.z); M4[3] = fmaxf(M4[3], v.w);
  }
#pragma unroll
  for (int m = 1; m <= 16; m <<= 1)
#pragma unroll
    for (int hh = 0; hh < 4; ++hh)
      M4[hh] = fmaxf(M4[hh], __shfl_xor(M4[hh], m, 32));

  float slc[8][4], mneg[8][4];
#pragma unroll
  for (int r = 0; r < 8; ++r) {
    const float4 v = *(const float4*)(sl + ((size_t)(b * N + i0 + half * 8 + r)) * 4);
    slc[r][0] = v.x; slc[r][1] = v.y; slc[r][2] = v.z; slc[r][3] = v.w;
#pragma unroll
    for (int hh = 0; hh < 4; ++hh) mneg[r][hh] = -lrelu(slc[r][hh] + M4[hh]);
  }

  // A fragments: normalized query features (reused for all 64 j-chunks).
  const __bf16* nfi = nfb + (size_t)(i0 + row) * C;
  v16bf af[4];
#pragma unroll
  for (int kc = 0; kc < 4; ++kc) {
    const v8bf lo = *(const v8bf*)(nfi + kc * 32 + half * 8);
    const v8bf hi = *(const v8bf*)(nfi + kc * 32 + 16 + half * 8);
    af[kc] = cat16(lo, hi);
  }

  v8f acc[4][2];
  const v8f vz = {};
#pragma unroll
  for (int hh = 0; hh < 4; ++hh) { acc[hh][0] = vz; acc[hh][1] = vz; }
  float ls[4][8];
#pragma unroll
  for (int hh = 0; hh < 4; ++hh)
#pragma unroll
    for (int r = 0; r < 8; ++r) ls[hh][r] = 0.f;

  issue(0, 0);                                          // prologue: panel 0

  for (int k = 0; k < N / 32; ++k) {
    const int j0 = k * 32;
    issue((j0 + 32) & (N - 1), (k + 1) & 1);            // prefetch next panel
    asm volatile("s_wait_asynccnt 8" ::: "memory");     // current panel landed
    __syncthreads();                                    // ... for all waves
    const __bf16* nfc = nfbuf[k & 1];
    const __bf16* htc = htbuf[k & 1];

    v8f sim[2];
    float sr4[2][4];
#pragma unroll
    for (int jt = 0; jt < 2; ++jt) {
      const __bf16* nfj = nfc + (size_t)(jt * 16 + row) * C;  // B[k][j]=nf[j][k]
      v8f d = {};
#pragma unroll
      for (int kc = 0; kc < 4; ++kc) {
        const v8bf lo = *(const v8bf*)(nfj + kc * 32 + half * 8);
        const v8bf hi = *(const v8bf*)(nfj + kc * 32 + half * 8 + 16);
        d = WMMA_BF16(af[kc], cat16(lo, hi), d);
      }
      sim[jt] = d;
      const float4 v = *(const float4*)(srb + (size_t)(j0 + jt * 16 + row) * 4);
      sr4[jt][0] = v.x; sr4[jt][1] = v.y; sr4[jt][2] = v.z; sr4[jt][3] = v.w;
    }

    // p = mask ? exp(leaky(s_l+s_r) - m) : 0, to LDS (D->A lane transpose).
#pragma unroll
    for (int hh = 0; hh < 4; ++hh)
#pragma unroll
      for (int jt = 0; jt < 2; ++jt) {
        const float srv = sr4[jt][hh];
#pragma unroll
        for (int r = 0; r < 8; ++r) {
          const float lg = lrelu(slc[r][hh] + srv);
          const float p  = (sim[jt][r] > 0.f) ? __expf(lg + mneg[r][hh]) : 0.f;
          ls[hh][r] += p;
          stage[wv][hh][(half * 8 + r) * 32 + jt * 16 + row] = (__bf16)p;
        }
      }
    asm volatile("s_wait_dscnt 0" ::: "memory");

    // out += P * h^T  (2 WMMAs per head: c split 16+16).
#pragma unroll
    for (int hh = 0; hh < 4; ++hh) {
      const __bf16* sp = &stage[wv][hh][row * 32 + half * 8];
      const v8bf plo = *(const v8bf*)(sp);
      const v8bf phi = *(const v8bf*)(sp + 16);
      const v16bf pa = cat16(plo, phi);
#pragma unroll
      for (int t = 0; t < 2; ++t) {
        const __bf16* hp = htc + (size_t)(hh * 32 + t * 16 + row) * 32 + half * 8;
        const v8bf hlo = *(const v8bf*)(hp);
        const v8bf hhi = *(const v8bf*)(hp + 16);
        acc[hh][t] = WMMA_BF16(pa, cat16(hlo, hhi), acc[hh][t]);
      }
    }
    asm volatile("s_wait_dscnt 0" ::: "memory");        // reads done before
    __syncthreads();                                    // next panel overwrite
  }

  // Row sums of P (denominator): reduce across the 16-lane group.
#pragma unroll
  for (int m = 1; m <= 8; m <<= 1)
#pragma unroll
    for (int hh = 0; hh < 4; ++hh)
#pragma unroll
      for (int r = 0; r < 8; ++r) ls[hh][r] += __shfl_xor(ls[hh][r], m, 32);

  float* outb = out + ((size_t)(b * N + i0)) * C;
#pragma unroll
  for (int hh = 0; hh < 4; ++hh)
#pragma unroll
    for (int t = 0; t < 2; ++t)
#pragma unroll
      for (int r = 0; r < 8; ++r) {
        const float invl = 1.0f / fmaxf(ls[hh][r], 1e-30f);
        const int i = half * 8 + r;
        const int c = hh * 32 + t * 16 + row;
        outb[(size_t)i * C + c] = acc[hh][t][r] * invl;
      }
}

// ---------------------------------------------------------------------------
extern "C" void kernel_launch(void* const* d_in, const int* in_sizes, int n_in,
                              void* d_out, int out_size, void* d_ws, size_t ws_size,
                              hipStream_t stream) {
  const int B = 4, N = 2048, C = 128;
  const float* x    = (const float*)d_in[0];
  const float* W    = (const float*)d_in[1];
  const float* bias = (const float*)d_in[2];
  const float* a    = (const float*)d_in[3];
  float* out = (float*)d_out;

  char* ws = (char*)d_ws;
  __bf16* nf = (__bf16*)ws;                                     // B*N*C bf16 (2 MiB)
  __bf16* hT = (__bf16*)(ws + (size_t)B * N * C * 2);           // B*C*N bf16 (2 MiB)
  float*  sl = (float*)(ws + (size_t)B * N * C * 4);            // B*N*4 f32
  float*  sr = (float*)(ws + (size_t)B * N * C * 4 + (size_t)B * N * 16);

  gat_prep<<<B * N / 16, 32, 0, stream>>>(x, W, bias, a, nf, hT, sl, sr);
  gat_attn<<<B * N / 64, 128, 0, stream>>>(nf, hT, sl, sr, out);
}
